// Indexer_53626961658291
// MI455X (gfx1250) — 16-bit and 8-bit matrix operand order repaired and checked by reading, compiled, not run on hardware
//
#include <hip/hip_runtime.h>
#include <hip/hip_bf16.h>


#define T_  1024
#define HID_ 7168
#define R_  1536
#define H_  64
#define D_  128
#define HD_ 8192
#define TOPK_ 512
#define KT_ 64

typedef __attribute__((ext_vector_type(16))) __bf16 v16bf;
typedef __attribute__((ext_vector_type(8)))  __bf16 v8bf;
typedef __attribute__((ext_vector_type(8)))  float  v8f;
typedef __attribute__((ext_vector_type(16))) int    v16i;
typedef __attribute__((ext_vector_type(4)))  int    v4i;
typedef __attribute__((ext_vector_type(2)))  int    v2i;

#if defined(__has_builtin)
#if __has_builtin(__builtin_amdgcn_tensor_load_to_lds) && \
    __has_builtin(__builtin_amdgcn_s_wait_tensorcnt)
#define USE_TDM 1
#endif
#endif
#ifndef USE_TDM
#define USE_TDM 0
#endif


__device__ __forceinline__ unsigned char f32_to_e4m3(float x) {
  unsigned int u = __float_as_uint(x);
  unsigned char sgn = (unsigned char)((u >> 24) & 0x80u);
  float ax = __uint_as_float(u & 0x7FFFFFFFu);
  if (ax != ax) return (unsigned char)(sgn | 0x7F);
  if (ax >= 464.0f) return (unsigned char)(sgn | 0x7E);
  if (ax < 0.0009765625f) return sgn;
  int e;
  float m = frexpf(ax, &e);
  int E = e - 1 + 7;
  if (E >= 1) {
    int mant = (int)(m * 16.0f + 0.5f) - 8;
    if (mant >= 8) { mant = 0; E += 1; }
    if (E > 15 || (E == 15 && mant == 7)) return (unsigned char)(sgn | 0x7E);
    return (unsigned char)(sgn | (E << 3) | mant);
  }
  int mant = (int)(ax * 512.0f + 0.5f);
  if (mant > 7) mant = 7;
  return (unsigned char)(sgn | mant);
}

#if USE_TDM
typedef __attribute__((ext_vector_type(4))) unsigned int u32x4;
typedef __attribute__((ext_vector_type(8))) int          i32x8;
typedef __attribute__((ext_vector_type(4))) int          i32x4;

__device__ __forceinline__ void tdm_load_2d(unsigned lds_byte_off,
                                            const void* gptr,
                                            int k_elems, int tile_rows) {
  unsigned long long ga = (unsigned long long)(size_t)gptr;
  unsigned ks   = (unsigned)k_elems;
  unsigned rows = (unsigned)tile_rows;
  u32x4 g0 = {
      (unsigned)__builtin_amdgcn_readfirstlane((int)1u),
      (unsigned)__builtin_amdgcn_readfirstlane((int)lds_byte_off),
      (unsigned)__builtin_amdgcn_readfirstlane((int)(unsigned)ga),
      (unsigned)__builtin_amdgcn_readfirstlane(
          (int)((unsigned)(ga >> 32) | (2u << 30)))};
  i32x8 g1 = {
      __builtin_amdgcn_readfirstlane((int)(1u << 16)),
      __builtin_amdgcn_readfirstlane((int)((ks & 0xFFFFu) << 16)),
      __builtin_amdgcn_readfirstlane(
          (int)((ks >> 16) | ((rows & 0xFFFFu) << 16))),
      __builtin_amdgcn_readfirstlane(
          (int)((rows >> 16) | ((unsigned)KT_ << 16))),
      __builtin_amdgcn_readfirstlane((int)rows),
      __builtin_amdgcn_readfirstlane((int)ks),
      0, 0};
  i32x4 z4 = {0, 0, 0, 0};
  i32x8 z8 = {0, 0, 0, 0, 0, 0, 0, 0};
  __builtin_amdgcn_tensor_load_to_lds(g0, g1, z4, z4, z8, 0);
}
#endif


__global__ __launch_bounds__(256) void cvt_bf16_kernel(
    const float* __restrict__ src, __bf16* __restrict__ dst, int n) {
  int i = blockIdx.x * 256 + threadIdx.x;
  if (i < n) dst[i] = (__bf16)src[i];
}

__global__ __launch_bounds__(256) void cvt_bf16_t_kernel(
    const float* __restrict__ src, __bf16* __restrict__ dst, int rows, int cols) {
  int i = blockIdx.x * 256 + threadIdx.x;
  if (i < rows * cols) {
    int r = i / cols, c = i - r * cols;
    dst[(size_t)c * rows + r] = (__bf16)src[i];
  }
}


template <int BM, int BN>
__global__ __launch_bounds__(256) void gemm_bf16_wmma(
    const __bf16* __restrict__ A, const __bf16* __restrict__ Bt,
    float* __restrict__ C, int M, int N, int K) {
  __shared__ __align__(64) __bf16 smem[2][(BM + BN) * KT_];

  const int tid  = threadIdx.x;
  const int lane = tid & 31;
  const int wave = tid >> 5;
  const int half = lane >> 4;
  const int r    = lane & 15;

  const int nbn = N / BN;
  const int bm0 = (blockIdx.x / nbn) * BM;
  const int bn0 = (blockIdx.x % nbn) * BN;

  constexpr int WCOLS = BN / 32;
  const int wm0 = (wave / WCOLS) * 32;
  const int wn0 = (wave % WCOLS) * 32;

  v8f acc[2][2] = {};

  auto stage = [&](int b, int k0) {
#if USE_TDM
    unsigned base = (unsigned)b * (unsigned)sizeof(smem[0]);
    if (wave == 0) {
      tdm_load_2d(base, A + (size_t)bm0 * K + k0, K, BM);
    } else if (wave == 1) {
      tdm_load_2d(base + (unsigned)(BM * KT_ * 2), Bt + (size_t)bn0 * K + k0, K, BN);
    }
#else
    __bf16* As = smem[b];
    __bf16* Bs = smem[b] + BM * KT_;
    for (int idx = tid; idx < BM * (KT_ / 8); idx += 256) {
      int row = idx / (KT_ / 8), c = idx - row * (KT_ / 8);
      *(v8bf*)&As[row * KT_ + c * 8] =
          *(const v8bf*)(A + (size_t)(bm0 + row) * K + k0 + c * 8);
    }
    for (int idx = tid; idx < BN * (KT_ / 8); idx += 256) {
      int row = idx / (KT_ / 8), c = idx - row * (KT_ / 8);
      *(v8bf*)&Bs[row * KT_ + c * 8] =
          *(const v8bf*)(Bt + (size_t)(bn0 + row) * K + k0 + c * 8);
    }
#endif
  };

  const int nK = K / KT_;
  int buf = 0;

  stage(0, 0);
#if USE_TDM
  if (wave < 2) __builtin_amdgcn_s_wait_tensorcnt(0);
#endif
  __syncthreads();

  for (int kb = 0; kb < nK; ++kb) {
    if (kb + 1 < nK) stage(buf ^ 1, (kb + 1) * KT_);

    const __bf16* Ab = smem[buf];
    const __bf16* Bb = smem[buf] + BM * KT_;
    for (int kk = 0; kk < KT_; kk += 32) {
      v16bf af[2], bfr[2];
      for (int i = 0; i < 2; ++i) {
        const __bf16* ap = Ab + (size_t)(wm0 + i * 16 + r) * KT_ + kk;
        union { v16bf v; v8bf h[2]; } ua;
        ua.h[0] = *(const v8bf*)(ap + half * 8);
        ua.h[1] = *(const v8bf*)(ap + 16 + half * 8);
        af[i] = ua.v;
        const __bf16* bp = Bb + (size_t)(wn0 + i * 16 + r) * KT_ + kk;
        union { v16bf v; v8bf h[2]; } ub;
        ub.h[0] = *(const v8bf*)(bp + half * 8);
        ub.h[1] = *(const v8bf*)(bp + 16 + half * 8);
        bfr[i] = ub.v;
      }
      for (int i = 0; i < 2; ++i)
        for (int j = 0; j < 2; ++j)
          acc[i][j] = __builtin_amdgcn_wmma_f32_16x16x32_bf16(
              false, af[i], false, bfr[j], (short)0, acc[i][j], false, false);
    }

#if USE_TDM
    if (wave < 2) __builtin_amdgcn_s_wait_tensorcnt(0);
#endif
    __syncthreads();
    buf ^= 1;
  }

  for (int i = 0; i < 2; ++i)
    for (int j = 0; j < 2; ++j)
      for (int rr = 0; rr < 8; ++rr)
        C[(size_t)(bm0 + wm0 + i * 16 + rr + half * 8) * N +
          (bn0 + wn0 + j * 16 + r)] = acc[i][j][rr];
}


__global__ __launch_bounds__(128) void q_post_kernel(
    const float* __restrict__ qfull,
    const int*   __restrict__ pos,
    const float* __restrict__ wout,
    unsigned char* __restrict__ qf8,
    float* __restrict__ wts) {
  __shared__ float xs[128];
  __shared__ float red[128];
  const int row = blockIdx.x;
  const int t = row >> 6;
  const int d = threadIdx.x;

  float x = qfull[(size_t)t * HD_ + (size_t)(row & 63) * D_ + d];
  xs[d] = x;
  __syncthreads();

  float y = x;
  if (d < 64) {
    int j = (d < 32) ? d : (d - 32);
    float ang = (float)pos[t] * exp2f(-(2.0f * j / 64.0f) * 13.287712379549449f);
    float c = cosf(ang), s = sinf(ang);
    y = (d < 32) ? (xs[2 * j] * c - xs[2 * j + 1] * s)
                 : (xs[2 * j + 1] * c + xs[2 * j] * s);
  }
  __syncthreads();
  xs[d] = y;
  __syncthreads();

  for (int m = 1; m < 128; m <<= 1) {
    float a = xs[d], b = xs[d ^ m];
    __syncthreads();
    xs[d] = (d & m) ? (b - a) : (a + b);
    __syncthreads();
  }
  float v = xs[d] * 0.08838834764831845f;

  red[d] = fabsf(v);
  __syncthreads();
  for (int s = 64; s > 0; s >>= 1) {
    if (d < s) red[d] = fmaxf(red[d], red[d + s]);
    __syncthreads();
  }
  float amax = fmaxf(red[0], 1e-4f);
  float sf = exp2f(ceilf(log2f(amax * (1.0f / 448.0f))));
  qf8[(size_t)row * D_ + d] = f32_to_e4m3(v / sf);
  if (d == 0)
    wts[row] = wout[row] * 0.125f * sf * 0.08838834764831845f;
}


__global__ __launch_bounds__(128) void k_post_kernel(
    const float* __restrict__ kpre,
    const int*   __restrict__ pos,
    const float* __restrict__ gamma, const float* __restrict__ beta,
    unsigned char* __restrict__ kf8,
    float* __restrict__ kscale) {
  __shared__ float xs[128];
  __shared__ float red[128];
  const int t = blockIdx.x;
  const int d = threadIdx.x;

  float x = kpre[(size_t)t * D_ + d];
  red[d] = x;
  __syncthreads();
  for (int s = 64; s > 0; s >>= 1) {
    if (d < s) red[d] += red[d + s];
    __syncthreads();
  }
  float mu = red[0] * (1.0f / 128.0f);
  __syncthreads();
  float cz = x - mu;
  red[d] = cz * cz;
  __syncthreads();
  for (int s = 64; s > 0; s >>= 1) {
    if (d < s) red[d] += red[d + s];
    __syncthreads();
  }
  float var = red[0] * (1.0f / 128.0f);
  float xn = cz * rsqrtf(var + 1e-5f) * gamma[d] + beta[d];

  __syncthreads();
  xs[d] = xn;
  __syncthreads();

  float y = xn;
  if (d < 64) {
    int j = (d < 32) ? d : (d - 32);
    float ang = (float)pos[t] * exp2f(-(2.0f * j / 64.0f) * 13.287712379549449f);
    float c = cosf(ang), s = sinf(ang);
    y = (d < 32) ? (xs[2 * j] * c - xs[2 * j + 1] * s)
                 : (xs[2 * j + 1] * c + xs[2 * j] * s);
  }
  __syncthreads();
  xs[d] = y;
  __syncthreads();

  for (int m = 1; m < 128; m <<= 1) {
    float a = xs[d], b = xs[d ^ m];
    __syncthreads();
    xs[d] = (d & m) ? (b - a) : (a + b);
    __syncthreads();
  }
  float v = xs[d] * 0.08838834764831845f;

  red[d] = fabsf(v);
  __syncthreads();
  for (int s = 64; s > 0; s >>= 1) {
    if (d < s) red[d] = fmaxf(red[d], red[d + s]);
    __syncthreads();
  }
  float amax = fmaxf(red[0], 1e-4f);
  float sf = exp2f(ceilf(log2f(amax * (1.0f / 448.0f))));
  kf8[(size_t)t * D_ + d] = f32_to_e4m3(v / sf);
  if (d == 0) kscale[t] = sf;
}


__global__ __launch_bounds__(256) void scores_topk_kernel(
    const unsigned char* __restrict__ qf8,
    const unsigned char* __restrict__ kf8,
    const float* __restrict__ kscale,
    const float* __restrict__ wts,
    float* __restrict__ vals_out,
    int*   __restrict__ idx_out) {
  __shared__ float s_score[1024];
  __shared__ int   s_idx[1024];

  const int t    = blockIdx.x;
  const int lane = threadIdx.x & 31;
  const int wave = threadIdx.x >> 5;
  const int half = lane >> 4;
  const int r    = lane & 15;

  v16i afrag[4];
  const unsigned char* qbase = qf8 + (size_t)t * H_ * D_;
  for (int mt = 0; mt < 4; ++mt) {
    const unsigned char* ap = qbase + (size_t)(mt * 16 + r) * D_ + half * 8;
    union { v16i v; v2i c[8]; } ua;
    for (int c = 0; c < 8; ++c) ua.c[c] = *(const v2i*)(ap + 16 * c);
    afrag[mt] = ua.v;
  }
  float wv[4][8];
  for (int mt = 0; mt < 4; ++mt)
    for (int rr = 0; rr < 8; ++rr)
      wv[mt][rr] = wts[t * H_ + mt * 16 + rr + half * 8];

  for (int nt = 0; nt < 8; ++nt) {
    const int s = wave * 128 + nt * 16 + r;
    const unsigned char* bp = kf8 + (size_t)s * D_ + half * 8;
    union { v16i v; v2i c[8]; } ub;
    for (int c = 0; c < 8; ++c) ub.c[c] = *(const v2i*)(bp + 16 * c);

    float partial = 0.0f;
    for (int mt = 0; mt < 4; ++mt) {
      v8f acc = {};
      acc = __builtin_amdgcn_wmma_f32_16x16x128_fp8_fp8(
          afrag[mt], ub.v, (short)0, acc, false, false);
      for (int rr = 0; rr < 8; ++rr)
        partial += wv[mt][rr] * fmaxf(acc[rr], 0.0f);
    }
    partial += __shfl_xor(partial, 16, 32);
    float sc = partial * kscale[s];
    if (s > t) sc = -1e30f;
    if (half == 0) { s_score[s] = sc; s_idx[s] = s; }
  }
  __syncthreads();

  for (int kk = 2; kk <= 1024; kk <<= 1) {
    for (int j = kk >> 1; j > 0; j >>= 1) {
      for (int i = threadIdx.x; i < 1024; i += 256) {
        int ixj = i ^ j;
        if (ixj > i) {
          bool desc = ((i & kk) == 0);
          float a = s_score[i], b = s_score[ixj];
          if (desc ? (a < b) : (a > b)) {
            s_score[i] = b; s_score[ixj] = a;
            int ti = s_idx[i]; s_idx[i] = s_idx[ixj]; s_idx[ixj] = ti;
          }
        }
      }
      __syncthreads();
    }
  }

  for (int i = threadIdx.x; i < TOPK_; i += 256) {
    vals_out[(size_t)t * TOPK_ + i] = s_score[i];
    idx_out[(size_t)t * TOPK_ + i]  = s_idx[i];
  }
}


extern "C" void kernel_launch(void* const* d_in, const int* in_sizes, int n_in,
                              void* d_out, int out_size, void* d_ws, size_t ws_size,
                              hipStream_t stream) {
  const float* hid   = (const float*)d_in[0];
  const float* qlora = (const float*)d_in[1];
  const int*   pos   = (const int*)d_in[2];
  const float* wqb   = (const float*)d_in[3];
  const float* wk    = (const float*)d_in[4];
  const float* wproj = (const float*)d_in[5];
  const float* lng   = (const float*)d_in[6];
  const float* lnb   = (const float*)d_in[7];

  char* base = (char*)d_ws;
  size_t off = 0;
  auto alloc = [&](size_t bytes) -> void* {
    void* p = base + off;
    off = (off + bytes + 255) & ~(size_t)255;
    return p;
  };

  __bf16* qlora_bf = (__bf16*)alloc((size_t)T_ * R_ * 2);
  __bf16* hid_bf   = (__bf16*)alloc((size_t)T_ * HID_ * 2);
  __bf16* wqbT     = (__bf16*)alloc((size_t)HD_ * R_ * 2);
  __bf16* wkT      = (__bf16*)alloc((size_t)D_ * HID_ * 2);
  __bf16* wpT      = (__bf16*)alloc((size_t)H_ * HID_ * 2);
  float*  qfull    = (float*)alloc((size_t)T_ * HD_ * 4);
  float*  kpre     = (float*)alloc((size_t)T_ * D_ * 4);
  float*  wout     = (float*)alloc((size_t)T_ * H_ * 4);
  unsigned char* qf8 = (unsigned char*)alloc((size_t)T_ * H_ * D_);
  unsigned char* kf8 = (unsigned char*)alloc((size_t)T_ * D_);
  float*  kscale   = (float*)alloc((size_t)T_ * 4);
  float*  wts      = (float*)alloc((size_t)T_ * H_ * 4);

  {
    int n = T_ * R_;
    cvt_bf16_kernel<<<(n + 255) / 256, 256, 0, stream>>>(qlora, qlora_bf, n);
    n = T_ * HID_;
    cvt_bf16_kernel<<<(n + 255) / 256, 256, 0, stream>>>(hid, hid_bf, n);
    n = R_ * HD_;
    cvt_bf16_t_kernel<<<(n + 255) / 256, 256, 0, stream>>>(wqb, wqbT, R_, HD_);
    n = HID_ * D_;
    cvt_bf16_t_kernel<<<(n + 255) / 256, 256, 0, stream>>>(wk, wkT, HID_, D_);
    n = HID_ * H_;
    cvt_bf16_t_kernel<<<(n + 255) / 256, 256, 0, stream>>>(wproj, wpT, HID_, H_);
  }

  gemm_bf16_wmma<64, 128><<<(T_ / 64) * (HD_ / 128), 256, 0, stream>>>(
      qlora_bf, wqbT, qfull, T_, HD_, R_);
  gemm_bf16_wmma<64, 128><<<(T_ / 64) * (D_ / 128), 256, 0, stream>>>(
      hid_bf, wkT, kpre, T_, D_, HID_);
  gemm_bf16_wmma<128, 64><<<(T_ / 128) * (H_ / 64), 256, 0, stream>>>(
      hid_bf, wpT, wout, T_, H_, HID_);

  k_post_kernel<<<T_, 128, 0, stream>>>(kpre, pos, lng, lnb, kf8, kscale);
  q_post_kernel<<<T_ * H_, 128, 0, stream>>>(qfull, pos, wout, qf8, wts);

  float* vals = (float*)d_out;
  int*   idxo = (int*)d_out + (size_t)T_ * TOPK_;
  scores_topk_kernel<<<T_, 256, 0, stream>>>(qf8, kf8, kscale, wts, vals, idxo);
}
